// MultiHeadAttention_47038481826429
// MI455X (gfx1250) — matrix operand order repaired and checked by reading, not compiled, not run on hardware
//
#include <hip/hip_runtime.h>

typedef __bf16 bf16_t;
typedef bf16_t v16bf __attribute__((ext_vector_type(16)));
typedef float  v8f   __attribute__((ext_vector_type(8)));
typedef int    v4i_  __attribute__((__vector_size__(16)));

// ---------- WMMA fragment loaders (16-bit, contiguous k = 16h+s order used by A and B alike, wave32) ----------

// A-matrix 16x32 (MxK): lane L -> row = L&15; kbase = (L>>4)*16;
// elements 0..15 = K kbase..kbase+15 contiguous.
__device__ __forceinline__ v16bf load_frag_a(const bf16_t* base, int ld) {
  int lane = threadIdx.x & 31;
  int row  = lane & 15;
  int kb   = (lane >> 4) << 4;
  const bf16_t* p = base + (size_t)row * ld + kb;
  union { uint4 u[2]; v16bf v; } t;
  t.u[0] = *reinterpret_cast<const uint4*>(p);
  t.u[1] = *reinterpret_cast<const uint4*>(p + 8);
  return t.v;
}

// B-matrix 32x16 (KxN), source given column-major (i.e. BT[n][k] row-major):
// lane L -> col = L&15; K = (L>>4)*16 .. +15 contiguous.
__device__ __forceinline__ v16bf load_frag_b(const bf16_t* baseT, int ld) {
  int lane = threadIdx.x & 31;
  int col  = lane & 15;
  int ks   = (lane >> 4) << 4;
  const bf16_t* p = baseT + (size_t)col * ld + ks;
  union { uint4 u[2]; v16bf v; } t;
  t.u[0] = *reinterpret_cast<const uint4*>(p);
  t.u[1] = *reinterpret_cast<const uint4*>(p + 8);
  return t.v;
}

__device__ __forceinline__ v8f wmma_bf16(v16bf a, v16bf b, v8f c) {
  return __builtin_amdgcn_wmma_f32_16x16x32_bf16(false, a, false, b, (short)0, c,
                                                 false, false);
}

// ---------- CDNA5 async global->LDS copy (16B), ASYNCcnt-tracked ----------

__device__ __forceinline__ void async_copy16(const bf16_t* g, bf16_t* l) {
#if __has_builtin(__builtin_amdgcn_global_load_async_to_lds_b128)
  __builtin_amdgcn_global_load_async_to_lds_b128(
      (v4i_*)(g),
      (__attribute__((address_space(3))) v4i_*)(l), 0, 0);
#else
  *reinterpret_cast<uint4*>(l) = *reinterpret_cast<const uint4*>(g);
#endif
}

__device__ __forceinline__ void wait_async0() {
#if __has_builtin(__builtin_amdgcn_s_wait_asynccnt)
  __builtin_amdgcn_s_wait_asynccnt(0);
#else
  asm volatile("s_wait_asynccnt 0" ::: "memory");
#endif
}

// ---------- conversion kernels ----------

__global__ void __launch_bounds__(256) f32_to_bf16_kernel(
    const float* __restrict__ src, bf16_t* __restrict__ dst, int n4) {
  int i = blockIdx.x * blockDim.x + threadIdx.x;
  if (i >= n4) return;
  float4 f = reinterpret_cast<const float4*>(src)[i];
  union { bf16_t b[4]; uint2 u; } p;
  p.b[0] = (bf16_t)f.x; p.b[1] = (bf16_t)f.y;
  p.b[2] = (bf16_t)f.z; p.b[3] = (bf16_t)f.w;
  reinterpret_cast<uint2*>(dst)[i] = p.u;
}

// 1024x1024 f32 -> bf16 transpose (dst[n][k] = src[k][n])
__global__ void __launch_bounds__(256) transpose_f32_to_bf16(
    const float* __restrict__ src, bf16_t* __restrict__ dst) {
  __shared__ float tile[32][33];
  int x = blockIdx.x * 32 + threadIdx.x;
  int y = blockIdx.y * 32 + threadIdx.y;
#pragma unroll
  for (int i = 0; i < 32; i += 8)
    tile[threadIdx.y + i][threadIdx.x] = src[(size_t)(y + i) * 1024 + x];
  __syncthreads();
  int x2 = blockIdx.y * 32 + threadIdx.x;
  int y2 = blockIdx.x * 32 + threadIdx.y;
#pragma unroll
  for (int i = 0; i < 32; i += 8)
    dst[(size_t)(y2 + i) * 1024 + x2] = (bf16_t)tile[threadIdx.x][threadIdx.y + i];
}

// ---------- bf16 GEMM: out = A[MxK] * BT[NxK]^T ----------
// Block = 8 waves = 128 rows x 64 cols. Shared B tile (64x32 bf16 = 4KB) is
// async-staged into LDS (double buffered) and reused by all 8 waves.
// mode 0: bf16 out, *scale          (Q/K projections)
// mode 1: bf16 out transposed per head -> [B,H,HD,S]   (V projection)
// mode 2: f32 out + bias            (final projection)
__global__ void __launch_bounds__(256) gemm_bf16_kernel(
    const bf16_t* __restrict__ A, const bf16_t* __restrict__ BT,
    void* __restrict__ outp, const float* __restrict__ bias,
    int M, int N, int K, float scale, int mode) {
  __shared__ __align__(16) bf16_t btile[2][64 * 32];
  int w = threadIdx.x >> 5, lane = threadIdx.x & 31;
  int row0 = blockIdx.y * 128 + w * 16;
  int col0 = blockIdx.x * 64;

  // cooperative stage of one 64x32 B tile: thread t copies 16B
  int tcol = threadIdx.x >> 2;
  int tko  = (threadIdx.x & 3) << 3;
  const bf16_t* bsrc0 = BT + (size_t)(col0 + tcol) * K + tko;
  bf16_t* ldst0 = &btile[0][tcol * 32 + tko];
  bf16_t* ldst1 = &btile[1][tcol * 32 + tko];

  v8f acc[4] = {};
  async_copy16(bsrc0, ldst0);  // prologue: stage k0 = 0
  for (int k0 = 0; k0 < K; k0 += 32) {
    int cur = (k0 >> 5) & 1;
    wait_async0();
    __syncthreads();  // tile[cur] visible to all 8 waves
    if (k0 + 32 < K)  // overlap next stage with this chunk's WMMAs
      async_copy16(bsrc0 + k0 + 32, cur ? ldst0 : ldst1);
    v16bf a = load_frag_a(A + (size_t)row0 * K + k0, K);
#pragma unroll
    for (int j = 0; j < 4; ++j) {
      v16bf b = load_frag_b(&btile[cur][j * 16 * 32], 32);
      acc[j] = wmma_bf16(a, b, acc[j]);
    }
  }

  int rbase = row0 + ((lane >> 4) << 3);
  int cl = lane & 15;
#pragma unroll
  for (int j = 0; j < 4; ++j) {
    int col = col0 + 16 * j + cl;
#pragma unroll
    for (int r = 0; r < 8; ++r) {
      int row = rbase + r;
      float v = acc[j][r] * scale;
      if (mode == 0) {
        ((bf16_t*)outp)[(size_t)row * N + col] = (bf16_t)v;
      } else if (mode == 1) {
        int b_ = row >> 11, s_ = row & 2047;   // S = 2048
        int h_ = col >> 6, d_ = col & 63;      // HD = 64
        ((bf16_t*)outp)[((size_t)(b_ * 16 + h_) * 64 + d_) * 2048 + s_] = (bf16_t)v;
      } else {
        ((float*)outp)[(size_t)row * N + col] = v + bias[col];
      }
    }
  }
}

// ---------- flash attention: one 16-query tile per wave ----------
// Q,K: [B*S, D] bf16 (Q pre-scaled by 1/8);  Vt: [B,H,HD,S] bf16; ctx: [B*S, D] bf16
__global__ void __launch_bounds__(256) attn_kernel(
    const bf16_t* __restrict__ Q, const bf16_t* __restrict__ Km,
    const bf16_t* __restrict__ Vt, bf16_t* __restrict__ ctx) {
  __shared__ __align__(16) bf16_t plds[8][16][32];  // per-wave P staging
  int w = threadIdx.x >> 5, lane = threadIdx.x & 31;
  int grp = blockIdx.x & 15;   // 16 groups of 8 query tiles
  int bh  = blockIdx.x >> 4;
  int h = bh & 15, b = bh >> 4;
  int q0 = (grp * 8 + w) * 16;

  const bf16_t* qptr = Q + (size_t)(b * 2048 + q0) * 1024 + h * 64;
  v16bf aQ0 = load_frag_a(qptr, 1024);
  v16bf aQ1 = load_frag_a(qptr + 32, 1024);

  float m[8], l[8];
  v8f acc[4] = {};
#pragma unroll
  for (int r = 0; r < 8; ++r) { m[r] = -1e30f; l[r] = 0.f; }

  for (int kb = 0; kb < q0 + 16; kb += 32) {
    v8f z = {};
    const bf16_t* kp0 = Km + (size_t)(b * 2048 + kb) * 1024 + h * 64;
    const bf16_t* kp1 = kp0 + (size_t)16 * 1024;
    if (kb + 32 < q0 + 16)  // speculative prefetch of next K chunk
      __builtin_prefetch(kp0 + (size_t)32 * 1024, 0, 1);
    v8f s0 = wmma_bf16(aQ0, load_frag_b(kp0, 1024), z);
    s0 = wmma_bf16(aQ1, load_frag_b(kp0 + 32, 1024), s0);
    v8f s1 = wmma_bf16(aQ0, load_frag_b(kp1, 1024), z);
    s1 = wmma_bf16(aQ1, load_frag_b(kp1 + 32, 1024), s1);

    if (kb + 31 > q0) {  // causal mask (key > query -> -inf)
      int colb = kb + (lane & 15);
      int rowb = q0 + ((lane >> 4) << 3);
#pragma unroll
      for (int r = 0; r < 8; ++r) {
        int row = rowb + r;
        if (colb > row)      s0[r] = -1e30f;
        if (colb + 16 > row) s1[r] = -1e30f;
      }
    }

#pragma unroll
    for (int r = 0; r < 8; ++r) {
      float v = fmaxf(s0[r], s1[r]);
      v = fmaxf(v, __shfl_xor(v, 8, 16));
      v = fmaxf(v, __shfl_xor(v, 4, 16));
      v = fmaxf(v, __shfl_xor(v, 2, 16));
      v = fmaxf(v, __shfl_xor(v, 1, 16));
      float mn = fmaxf(m[r], v);
      float al = __expf(m[r] - mn);
      m[r] = mn;
      float p0 = __expf(s0[r] - mn);
      float p1 = __expf(s1[r] - mn);
      s0[r] = p0; s1[r] = p1;
      float rs = p0 + p1;
      rs += __shfl_xor(rs, 8, 16);
      rs += __shfl_xor(rs, 4, 16);
      rs += __shfl_xor(rs, 2, 16);
      rs += __shfl_xor(rs, 1, 16);
      l[r] = l[r] * al + rs;
#pragma unroll
      for (int j = 0; j < 4; ++j) acc[j][r] *= al;
    }

    // stage P (bf16) in LDS, reload in A-fragment layout
    {
      int prow = (lane >> 4) << 3;
      int pcol = lane & 15;
#pragma unroll
      for (int r = 0; r < 8; ++r) {
        plds[w][prow + r][pcol]      = (bf16_t)s0[r];
        plds[w][prow + r][pcol + 16] = (bf16_t)s1[r];
      }
    }
    asm volatile("s_wait_dscnt 0" ::: "memory");
    v16bf aP = load_frag_a(&plds[w][0][0], 32);

    const bf16_t* vb = Vt + (size_t)(b * 16 + h) * 64 * 2048 + kb;
#pragma unroll
    for (int j = 0; j < 4; ++j) {
      v16bf bV = load_frag_b(vb + (size_t)(16 * j) * 2048, 2048);
      acc[j] = wmma_bf16(aP, bV, acc[j]);
    }
    asm volatile("s_wait_dscnt 0" ::: "memory");  // LDS WAR fence before next chunk
  }

  int rbase = q0 + ((lane >> 4) << 3);
  int cl = lane & 15;
#pragma unroll
  for (int j = 0; j < 4; ++j) {
#pragma unroll
    for (int r = 0; r < 8; ++r) {
      float v = acc[j][r] / l[r];
      ctx[(size_t)(b * 2048 + rbase + r) * 1024 + h * 64 + 16 * j + cl] = (bf16_t)v;
    }
  }
}

// ---------- launch ----------

extern "C" void kernel_launch(void* const* d_in, const int* in_sizes, int n_in,
                              void* d_out, int out_size, void* d_ws, size_t ws_size,
                              hipStream_t stream) {
  (void)in_sizes; (void)n_in; (void)out_size; (void)ws_size;
  const float* x  = (const float*)d_in[0];
  const float* Wq = (const float*)d_in[1];
  const float* Wk = (const float*)d_in[2];
  const float* Wv = (const float*)d_in[3];
  const float* Wo = (const float*)d_in[4];
  const float* bo = (const float*)d_in[5];
  float* out = (float*)d_out;

  char* ws = (char*)d_ws;
  const size_t MB = 1u << 20;
  bf16_t* xb  = (bf16_t*)(ws);             // 8 MB  [B*S, D]
  bf16_t* wqT = (bf16_t*)(ws + 8  * MB);   // 2 MB each, transposed [N,K]
  bf16_t* wkT = (bf16_t*)(ws + 10 * MB);
  bf16_t* wvT = (bf16_t*)(ws + 12 * MB);
  bf16_t* woT = (bf16_t*)(ws + 14 * MB);
  bf16_t* Qm  = (bf16_t*)(ws + 16 * MB);   // 8 MB [B*S, D] (scaled 1/8)
  bf16_t* Km  = (bf16_t*)(ws + 24 * MB);   // 8 MB [B*S, D]
  bf16_t* Vt  = (bf16_t*)(ws + 32 * MB);   // 8 MB [B,H,HD,S]
  bf16_t* ctx = (bf16_t*)(ws + 40 * MB);   // 8 MB [B*S, D]

  // 1) x -> bf16
  f32_to_bf16_kernel<<<4096, 256, 0, stream>>>(x, xb, (2 * 2048 * 1024) / 4);
  // 2) weights -> transposed bf16
  dim3 tb(32, 8), tg(32, 32);
  transpose_f32_to_bf16<<<tg, tb, 0, stream>>>(Wq, wqT);
  transpose_f32_to_bf16<<<tg, tb, 0, stream>>>(Wk, wkT);
  transpose_f32_to_bf16<<<tg, tb, 0, stream>>>(Wv, wvT);
  transpose_f32_to_bf16<<<tg, tb, 0, stream>>>(Wo, woT);
  // 3) QKV projections (M=4096, N=K=1024); Q pre-scaled by 1/sqrt(HD)=0.125
  dim3 gg(1024 / 64, 4096 / 128);
  gemm_bf16_kernel<<<gg, 256, 0, stream>>>(xb, wqT, Qm, nullptr, 4096, 1024, 1024, 0.125f, 0);
  gemm_bf16_kernel<<<gg, 256, 0, stream>>>(xb, wkT, Km, nullptr, 4096, 1024, 1024, 1.0f, 0);
  gemm_bf16_kernel<<<gg, 256, 0, stream>>>(xb, wvT, Vt, nullptr, 4096, 1024, 1024, 1.0f, 1);
  // 4) causal flash attention: B*H*(S/16)/8 blocks of 8 waves
  attn_kernel<<<2 * 16 * 16, 256, 0, stream>>>(Qm, Km, Vt, ctx);
  // 5) output projection + bias (f32 out)
  gemm_bf16_kernel<<<gg, 256, 0, stream>>>(ctx, woT, out, bo, 4096, 1024, 1024, 1.0f, 2);
}
